// SCRNNet_25357486916087
// MI455X (gfx1250) — matrix operand order repaired and checked by reading, not compiled, not run on hardware
//
#include <hip/hip_runtime.h>
#include <hip/hip_bf16.h>
#include <stdint.h>
#include <math.h>

// ---------------------------------------------------------------------------
// SCRN on MI455X (gfx1250, wave32, WMMA bf16 -> f32 accumulate)
// Pipeline:
//   0) f32 -> bf16 weight conversions (emb, W_x, U_c, V_h, fc_w)
//   1) wx = emb[x] @ W_x^T + b_x          (WMMA, fused gather, 32x64/wave)
//   2) c_all = linear scan over t          (parallel over 65536 chains)
//   3) u = c_all @ U_c^T                   (WMMA, hoisted, 32x64/wave)
//   4) persistent recurrence: h_t = tanh(u_t + h @ V_h^T)  (WMMA + device barrier)
//   5) out = h_last @ fc_w^T + fc_b        (WMMA, 32x64/wave)
// ---------------------------------------------------------------------------

typedef __bf16 bf16;
typedef __attribute__((ext_vector_type(16))) __bf16 v16bf;
typedef __attribute__((ext_vector_type(8)))  float  v8f;

union Frag16 { v16bf v; uint4 q[2]; };

#define VOCAB_SZ 32000
#define B_SZ    64
#define S_SZ    512
#define EMB_SZ  512
#define CTX_SZ  1024
#define HID_SZ  1024
#define NCLS_SZ 32000
#define M_ALL   (B_SZ * S_SZ)          // 32768
#define REC_BLOCKS 32                  // persistent grid (256 waves = 256 tiles)

// ---------------- utility kernels ----------------
__global__ void cvt_f32_bf16(const float* __restrict__ in, bf16* __restrict__ out, int n) {
  int i = blockIdx.x * blockDim.x + threadIdx.x;
  int stride = gridDim.x * blockDim.x;
  for (; i < n; i += stride) out[i] = (bf16)in[i];
}

__global__ void zero_u32(uint32_t* __restrict__ p, int n) {
  int i = blockIdx.x * blockDim.x + threadIdx.x;
  if (i < n) p[i] = 0u;
}

// ---------------- fragment helpers ----------------
// A fragment: row r = lane&15 of the 16-row tile; half = lane>>4.
// 16 bf16 per lane: K = half*16 .. half*16+15 (contiguous)
__device__ inline v16bf load_a_frag(const bf16* __restrict__ arow, int k0, int half) {
  Frag16 a;
  a.q[0] = *reinterpret_cast<const uint4*>(arow + k0 + half * 16);
  a.q[1] = *reinterpret_cast<const uint4*>(arow + k0 + half * 16 + 8);
  return a.v;
}
// B fragment: column n = lane&15 -> weight row (W is [N,K] row-major);
// 16 consecutive K starting at (lane>>4)*16.
__device__ inline v16bf load_b_frag(const bf16* __restrict__ brow, int k0) {
  Frag16 b;
  b.q[0] = *reinterpret_cast<const uint4*>(brow + k0);
  b.q[1] = *reinterpret_cast<const uint4*>(brow + k0 + 8);
  return b.v;
}

// ---------------- 1) fused embedding gather + input projection ----------------
// 32x64 macro-tile per wave (2 M-tiles x 4 N-tiles), K = 512.
// wx[m, n] = sum_k emb_b[tok[m], k] * Wx_b[n, k] + bx[n]
__global__ __launch_bounds__(256)
void embed_wx_kernel(const int* __restrict__ tok,       // [32768]
                     const bf16* __restrict__ emb_b,    // [32000,512]
                     const bf16* __restrict__ Wx_b,     // [1024,512]
                     const float* __restrict__ bx,      // [1024]
                     float* __restrict__ wx)            // [32768,1024]
{
  const int lane = threadIdx.x & 31;
  const int wave = threadIdx.x >> 5;
  const int tid  = blockIdx.x * 8 + wave;     // 0 .. 16383 macro-tiles
  const int mg   = tid >> 4;                  // 1024 row groups (32 rows)
  const int ng   = tid & 15;                  // 16 col groups (64 cols)
  const int r    = lane & 15;
  const int half = lane >> 4;
  const int m0 = mg * 32, n0 = ng * 64;

  const bf16* arow[2];
#pragma unroll
  for (int mi = 0; mi < 2; ++mi)
    arow[mi] = emb_b + (size_t)tok[m0 + mi * 16 + r] * EMB_SZ;
  const bf16* brow[4];
#pragma unroll
  for (int ni = 0; ni < 4; ++ni)
    brow[ni] = Wx_b + (size_t)(n0 + ni * 16 + r) * EMB_SZ + half * 16;

  v8f acc[2][4];
#pragma unroll
  for (int mi = 0; mi < 2; ++mi)
#pragma unroll
    for (int ni = 0; ni < 4; ++ni) {
      const float bias = bx[n0 + ni * 16 + r];
#pragma unroll
      for (int i = 0; i < 8; ++i) acc[mi][ni][i] = bias;
    }

  for (int k0 = 0; k0 < EMB_SZ; k0 += 32) {
    v16bf af[2], bf[4];
#pragma unroll
    for (int mi = 0; mi < 2; ++mi) af[mi] = load_a_frag(arow[mi], k0, half);
#pragma unroll
    for (int ni = 0; ni < 4; ++ni) bf[ni] = load_b_frag(brow[ni], k0);
#pragma unroll
    for (int mi = 0; mi < 2; ++mi)
#pragma unroll
      for (int ni = 0; ni < 4; ++ni)
        acc[mi][ni] = __builtin_amdgcn_wmma_f32_16x16x32_bf16(
            false, af[mi], false, bf[ni], (short)0, acc[mi][ni], false, false);
  }

#pragma unroll
  for (int mi = 0; mi < 2; ++mi)
#pragma unroll
    for (int ni = 0; ni < 4; ++ni) {
      float* orow = wx + (size_t)(m0 + mi * 16 + half * 8) * CTX_SZ + n0 + ni * 16 + r;
#pragma unroll
      for (int i = 0; i < 8; ++i) orow[(size_t)i * CTX_SZ] = acc[mi][ni][i];
    }
}

// ---------------- 2) context EMA scan (parallel over B*CTX chains) ----------------
__global__ void cscan_kernel(const float* __restrict__ wx,   // [B,S,CTX] f32
                             bf16* __restrict__ c_all)       // [B,S,CTX] bf16
{
  const int idx = blockIdx.x * blockDim.x + threadIdx.x;     // 0..65535
  const int b = idx >> 10, ci = idx & (CTX_SZ - 1);
  const float* wrow = wx    + (size_t)b * S_SZ * CTX_SZ + ci;
  bf16*       crow  = c_all + (size_t)b * S_SZ * CTX_SZ + ci;
  float c = 0.0f;
  for (int t = 0; t < S_SZ; ++t) {
    c = 0.05f * c + 0.95f * wrow[(size_t)t * CTX_SZ];
    crow[(size_t)t * CTX_SZ] = (bf16)c;
  }
}

// ---------------- 3) u = c_all @ U_c^T (hoisted, fully parallel) ----------------
// 32x64 macro-tile per wave, K = 1024.
__global__ __launch_bounds__(256)
void u_gemm_kernel(const bf16* __restrict__ c_all,   // [32768,1024]
                   const bf16* __restrict__ Uc_b,    // [1024,1024]
                   float* __restrict__ u)            // [32768,1024]
{
  const int lane = threadIdx.x & 31;
  const int wave = threadIdx.x >> 5;
  const int tid  = blockIdx.x * 8 + wave;   // 0 .. 16383
  const int mg   = tid >> 4;
  const int ng   = tid & 15;
  const int r    = lane & 15;
  const int half = lane >> 4;
  const int m0 = mg * 32, n0 = ng * 64;

  const bf16* arow[2];
#pragma unroll
  for (int mi = 0; mi < 2; ++mi)
    arow[mi] = c_all + (size_t)(m0 + mi * 16 + r) * CTX_SZ;
  const bf16* brow[4];
#pragma unroll
  for (int ni = 0; ni < 4; ++ni)
    brow[ni] = Uc_b + (size_t)(n0 + ni * 16 + r) * CTX_SZ + half * 16;

  v8f acc[2][4];
#pragma unroll
  for (int mi = 0; mi < 2; ++mi)
#pragma unroll
    for (int ni = 0; ni < 4; ++ni)
#pragma unroll
      for (int i = 0; i < 8; ++i) acc[mi][ni][i] = 0.0f;

  for (int k0 = 0; k0 < CTX_SZ; k0 += 32) {
    v16bf af[2], bf[4];
#pragma unroll
    for (int mi = 0; mi < 2; ++mi) af[mi] = load_a_frag(arow[mi], k0, half);
#pragma unroll
    for (int ni = 0; ni < 4; ++ni) bf[ni] = load_b_frag(brow[ni], k0);
#pragma unroll
    for (int mi = 0; mi < 2; ++mi)
#pragma unroll
      for (int ni = 0; ni < 4; ++ni)
        acc[mi][ni] = __builtin_amdgcn_wmma_f32_16x16x32_bf16(
            false, af[mi], false, bf[ni], (short)0, acc[mi][ni], false, false);
  }

#pragma unroll
  for (int mi = 0; mi < 2; ++mi)
#pragma unroll
    for (int ni = 0; ni < 4; ++ni) {
      float* orow = u + (size_t)(m0 + mi * 16 + half * 8) * HID_SZ + n0 + ni * 16 + r;
#pragma unroll
      for (int i = 0; i < 8; ++i) orow[(size_t)i * HID_SZ] = acc[mi][ni][i];
    }
}

// ---------------- 4) persistent recurrence: h_t = tanh(u_t + h V_h^T) ------------
// 32 blocks x 8 waves = 256 waves; wave owns one fixed 16x16 tile of h for all t
// (un-blocked on purpose: latency-bound, maximize per-step parallelism).
// Device-wide sync per step via per-step arrive counters (zeroed each launch).
__global__ __launch_bounds__(256)
void recurrence_kernel(const bf16* __restrict__ Vh_b,   // [1024,1024]
                       const float* __restrict__ u,     // [B,S,HID] f32
                       bf16* __restrict__ h0,           // [64,1024] (zero-init)
                       bf16* __restrict__ h1,           // [64,1024]
                       int* __restrict__ bar)           // [S_SZ] ints, zero-init
{
  const int lane = threadIdx.x & 31;
  const int wave = threadIdx.x >> 5;
  const int tid  = blockIdx.x * 8 + wave;   // 0..255
  const int mt   = tid >> 6;                // 4 row tiles (batch)
  const int nt   = tid & 63;                // 64 col tiles (hidden)
  const int r    = lane & 15;
  const int half = lane >> 4;
  const int m0 = mt * 16, n0 = nt * 16;

  const bf16* brow = Vh_b + (size_t)(n0 + r) * HID_SZ + half * 16;
  // u[b, t, n] with b = m0+half*8+i, n = n0+r
  const float* ubase = u + ((size_t)(m0 + half * 8) * S_SZ) * HID_SZ + n0 + r;

  for (int t = 0; t < S_SZ; ++t) {
    const bf16* hp = (t & 1) ? h1 : h0;
    bf16*       hn = (t & 1) ? h0 : h1;

    v8f acc;
#pragma unroll
    for (int i = 0; i < 8; ++i)
      acc[i] = ubase[(size_t)i * S_SZ * HID_SZ + (size_t)t * HID_SZ];

    const bf16* arow = hp + (size_t)(m0 + r) * HID_SZ;
    for (int k0 = 0; k0 < HID_SZ; k0 += 32) {
      v16bf a = load_a_frag(arow, k0, half);
      v16bf b = load_b_frag(brow, k0);
      acc = __builtin_amdgcn_wmma_f32_16x16x32_bf16(false, a, false, b,
                                                    (short)0, acc, false, false);
    }

    bf16* orow = hn + (size_t)(m0 + half * 8) * HID_SZ + n0 + r;
#pragma unroll
    for (int i = 0; i < 8; ++i) orow[(size_t)i * HID_SZ] = (bf16)tanhf(acc[i]);

    // device-wide step barrier: all 256 tiles written before anyone reads them
    __threadfence();
    __syncthreads();
    if (threadIdx.x == 0) {
      __hip_atomic_fetch_add(&bar[t], 1, __ATOMIC_RELEASE, __HIP_MEMORY_SCOPE_AGENT);
      while (__hip_atomic_load(&bar[t], __ATOMIC_ACQUIRE, __HIP_MEMORY_SCOPE_AGENT)
             < REC_BLOCKS) {
        __builtin_amdgcn_s_sleep(1);
      }
    }
    __syncthreads();
  }
}

// ---------------- 5) final classifier: out = h_last @ fc_w^T + fc_b -------------
// 32x64 macro-tile per wave: 2 M-groups x 500 N-groups = 1000 waves = 125 blocks.
__global__ __launch_bounds__(256)
void fc_kernel(const bf16* __restrict__ h,      // [64,1024]
               const bf16* __restrict__ fcw_b,  // [32000,1024]
               const float* __restrict__ fcb,   // [32000]
               float* __restrict__ out)         // [64,32000]
{
  const int lane = threadIdx.x & 31;
  const int wave = threadIdx.x >> 5;
  const int tid  = blockIdx.x * 8 + wave;   // 0..999
  const int mg   = tid & 1;                 // 2 row groups (32 rows)
  const int ng   = tid >> 1;                // 500 col groups (64 cols)
  const int r    = lane & 15;
  const int half = lane >> 4;
  const int m0 = mg * 32, n0 = ng * 64;

  const bf16* arow[2];
#pragma unroll
  for (int mi = 0; mi < 2; ++mi)
    arow[mi] = h + (size_t)(m0 + mi * 16 + r) * HID_SZ;
  const bf16* brow[4];
#pragma unroll
  for (int ni = 0; ni < 4; ++ni)
    brow[ni] = fcw_b + (size_t)(n0 + ni * 16 + r) * HID_SZ + half * 16;

  v8f acc[2][4];
#pragma unroll
  for (int mi = 0; mi < 2; ++mi)
#pragma unroll
    for (int ni = 0; ni < 4; ++ni) {
      const float bias = fcb[n0 + ni * 16 + r];
#pragma unroll
      for (int i = 0; i < 8; ++i) acc[mi][ni][i] = bias;
    }

  for (int k0 = 0; k0 < HID_SZ; k0 += 32) {
    v16bf af[2], bf[4];
#pragma unroll
    for (int mi = 0; mi < 2; ++mi) af[mi] = load_a_frag(arow[mi], k0, half);
#pragma unroll
    for (int ni = 0; ni < 4; ++ni) bf[ni] = load_b_frag(brow[ni], k0);
#pragma unroll
    for (int mi = 0; mi < 2; ++mi)
#pragma unroll
      for (int ni = 0; ni < 4; ++ni)
        acc[mi][ni] = __builtin_amdgcn_wmma_f32_16x16x32_bf16(
            false, af[mi], false, bf[ni], (short)0, acc[mi][ni], false, false);
  }

#pragma unroll
  for (int mi = 0; mi < 2; ++mi)
#pragma unroll
    for (int ni = 0; ni < 4; ++ni) {
      float* orow = out + (size_t)(m0 + mi * 16 + half * 8) * NCLS_SZ + n0 + ni * 16 + r;
#pragma unroll
      for (int i = 0; i < 8; ++i) orow[(size_t)i * NCLS_SZ] = acc[mi][ni][i];
    }
}

// ---------------------------------------------------------------------------
extern "C" void kernel_launch(void* const* d_in, const int* in_sizes, int n_in,
                              void* d_out, int out_size, void* d_ws, size_t ws_size,
                              hipStream_t stream) {
  const int*   x    = (const int*)  d_in[0];
  const float* emb  = (const float*)d_in[1];
  const float* W_x  = (const float*)d_in[2];
  const float* b_x  = (const float*)d_in[3];
  const float* U_c  = (const float*)d_in[4];
  const float* V_h  = (const float*)d_in[5];
  const float* fc_w = (const float*)d_in[6];
  const float* fc_b = (const float*)d_in[7];
  float* out = (float*)d_out;

  uint8_t* ws = (uint8_t*)d_ws;
  size_t off = 0;
  auto carve = [&](size_t bytes) -> void* {
    void* p = ws + off;
    off += (bytes + 255) & ~(size_t)255;
    return p;
  };

  bf16* emb_b = (bf16*)carve((size_t)VOCAB_SZ * EMB_SZ * sizeof(bf16));
  bf16* Wx_b  = (bf16*)carve((size_t)CTX_SZ * EMB_SZ * sizeof(bf16));
  bf16* Uc_b  = (bf16*)carve((size_t)HID_SZ * CTX_SZ * sizeof(bf16));
  bf16* Vh_b  = (bf16*)carve((size_t)HID_SZ * HID_SZ * sizeof(bf16));
  bf16* fcw_b = (bf16*)carve((size_t)NCLS_SZ * HID_SZ * sizeof(bf16));
  float* wx_u = (float*)carve((size_t)M_ALL * CTX_SZ * sizeof(float)); // wx, then aliased as u
  bf16* c_all = (bf16*)carve((size_t)M_ALL * CTX_SZ * sizeof(bf16));
  bf16* h0    = (bf16*)carve((size_t)B_SZ * HID_SZ * sizeof(bf16));
  bf16* h1    = (bf16*)carve((size_t)B_SZ * HID_SZ * sizeof(bf16));
  int*  bar   = (int*) carve((size_t)S_SZ * sizeof(int));

  // 0) weight conversions to bf16
  cvt_f32_bf16<<<2048, 256, 0, stream>>>(emb,  emb_b, VOCAB_SZ * EMB_SZ);
  cvt_f32_bf16<<<2048, 256, 0, stream>>>(W_x,  Wx_b,  CTX_SZ * EMB_SZ);
  cvt_f32_bf16<<<2048, 256, 0, stream>>>(U_c,  Uc_b,  HID_SZ * CTX_SZ);
  cvt_f32_bf16<<<2048, 256, 0, stream>>>(V_h,  Vh_b,  HID_SZ * HID_SZ);
  cvt_f32_bf16<<<2048, 256, 0, stream>>>(fc_w, fcw_b, NCLS_SZ * HID_SZ);

  // 1) wx = emb[x] @ W_x^T + b_x      (16384 macro-tiles / 8 waves per block)
  embed_wx_kernel<<<2048, 256, 0, stream>>>(x, emb_b, Wx_b, b_x, wx_u);

  // 2) c scan (65536 chains)
  cscan_kernel<<<256, 256, 0, stream>>>(wx_u, c_all);

  // 3) u = c_all @ U_c^T   (overwrites dead wx buffer)
  u_gemm_kernel<<<2048, 256, 0, stream>>>(c_all, Uc_b, wx_u);

  // 4) recurrence init (h0 = 0, barrier counters = 0) + persistent kernel
  zero_u32<<<(B_SZ * HID_SZ / 2 + 255) / 256, 256, 0, stream>>>(
      (uint32_t*)h0, B_SZ * HID_SZ / 2);
  zero_u32<<<(S_SZ + 255) / 256, 256, 0, stream>>>((uint32_t*)bar, S_SZ);
  recurrence_kernel<<<REC_BLOCKS, 256, 0, stream>>>(Vh_b, wx_u, h0, h1, bar);

  // 5) out = h_last @ fc_w^T + fc_b   (h_last lands in h0 after 512 steps)
  fc_kernel<<<125, 256, 0, stream>>>(h0, fcw_b, fc_b, out);
}
